// AblationAttention_30597347016989
// MI455X (gfx1250) — matrix operand order repaired and checked by reading, not compiled, not run on hardware
//
#include <hip/hip_runtime.h>
#include <hip/hip_bf16.h>

typedef __bf16 bf16;
typedef __attribute__((ext_vector_type(16))) __bf16 v16bf;
typedef __attribute__((ext_vector_type(8)))  __bf16 v8bf;
typedef __attribute__((ext_vector_type(8)))  float  v8f;

namespace {
constexpr int CB = 2, CS = 2048, CE = 2048, CH = 16, CKV = 8, CD = 128;
constexpr int CG = CH / CKV;
constexpr float CEPS = 1.1920929e-07f;        // float32 eps (torch RMSNorm default)
constexpr float CSCALE = 0.08838834764831845f; // 1/sqrt(128)
}

__device__ __forceinline__ v8f wmma_bf16(v16bf a, v16bf b, v8f c) {
  return __builtin_amdgcn_wmma_f32_16x16x32_bf16(false, a, false, b, (short)0, c,
                                                 false, false);
}

__device__ __forceinline__ v16bf pack16(v8bf lo, v8bf hi) {
  v16bf r;
#pragma unroll
  for (int e = 0; e < 8; ++e) { r[e] = lo[e]; r[e + 8] = hi[e]; }
  return r;
}

// Async global -> LDS copy (16B per lane). LDS dest = per-lane byte address
// (low 32 bits of the generic pointer ARE the LDS offset per the flat-address
// aperture rule). Tracked by ASYNCcnt.
__device__ __forceinline__ void async_ld_b128(void* lds_ptr, const void* gptr) {
  const unsigned lds_addr = (unsigned)(unsigned long long)lds_ptr;
  asm volatile("global_load_async_to_lds_b128 %0, %1, off"
               :: "v"(lds_addr), "v"((unsigned long long)gptr)
               : "memory");
}
__device__ __forceinline__ void wait_async0() {
  asm volatile("s_wait_asynccnt 0" ::: "memory");
}

__device__ __forceinline__ float half_max16(float v) {
#pragma unroll
  for (int m = 1; m <= 8; m <<= 1) v = fmaxf(v, __shfl_xor(v, m, 32));
  return v;  // reduced over each 16-lane half
}
__device__ __forceinline__ float half_sum16(float v) {
#pragma unroll
  for (int m = 1; m <= 8; m <<= 1) v += __shfl_xor(v, m, 32);
  return v;
}
__device__ __forceinline__ float full_sum32(float v) {
#pragma unroll
  for (int m = 1; m <= 16; m <<= 1) v += __shfl_xor(v, m, 32);
  return v;
}

// ---------------------------------------------------------------- cast f32->bf16
__global__ void cast_f32_bf16(const float* __restrict__ in, bf16* __restrict__ out,
                              int n) {
  int i = blockIdx.x * blockDim.x + threadIdx.x;
  int stride = gridDim.x * blockDim.x;
  for (; i < n; i += stride) out[i] = (bf16)in[i];
}

// -------------------------------------------------- WMMA GEMM: C = A * W^T + bias
// A: (M,K) bf16 row-major.  W: (N,K) bf16 row-major (so columns of B = rows of W).
// 128x128 output tile per block, 8 waves (4x2), each wave 32x64 = 2x4 WMMA frags.
// Double-buffered LDS, next tile streamed in via async-to-LDS while WMMAs run.
template <bool OUT_BF16>
__global__ __launch_bounds__(256) void gemm_bias(const bf16* __restrict__ A,
                                                 const bf16* __restrict__ W,
                                                 const float* __restrict__ bias,
                                                 void* __restrict__ outp,
                                                 int M, int N, int K) {
  __shared__ bf16 lds_a[2][128 * 40];  // 32 k-cols + 8 pad (16B-aligned rows)
  __shared__ bf16 lds_b[2][128 * 40];
  const int t = threadIdx.x;
  const int lane = t & 31, w = t >> 5;
  const int wave_m = w & 3, wave_n = w >> 2;
  const int nb = lane & 15, hl = lane >> 4;
  const int m_block = blockIdx.y, n_block = blockIdx.x;

  v8f acc[2][4] = {};

  const int lrow = t >> 1, lcol = (t & 1) * 16;
  const bf16* Ag = A + (size_t)(m_block * 128 + lrow) * K + lcol;
  const bf16* Wg = W + (size_t)(n_block * 128 + lrow) * K + lcol;

  auto issue_tile = [&](int buf, int k0) {
    bf16* la = &lds_a[buf][lrow * 40 + lcol];
    bf16* lb = &lds_b[buf][lrow * 40 + lcol];
    async_ld_b128(la,     Ag + k0);
    async_ld_b128(la + 8, Ag + k0 + 8);
    async_ld_b128(lb,     Wg + k0);
    async_ld_b128(lb + 8, Wg + k0 + 8);
  };

  issue_tile(0, 0);
  wait_async0();
  __syncthreads();

  for (int k0 = 0; k0 < K; k0 += 32) {
    const int cur = (k0 >> 5) & 1;
    if (k0 + 32 < K) {
      issue_tile(cur ^ 1, k0 + 32);           // overlap with WMMAs below
      __builtin_prefetch(Ag + k0 + 64, 0, 1); // global_prefetch_b8 further ahead
      __builtin_prefetch(Wg + k0 + 64, 0, 1);
    }

    v16bf afrag[2], bfrag[4];
#pragma unroll
    for (int i = 0; i < 2; ++i) {
      const int m = wave_m * 32 + i * 16 + nb;  // A: lane = M row
      v8bf lo = *(const v8bf*)&lds_a[cur][m * 40 + hl * 8];       // K 0-7 / 8-15
      v8bf hi = *(const v8bf*)&lds_a[cur][m * 40 + 16 + hl * 8];  // K 16-23 / 24-31
      afrag[i] = pack16(lo, hi);
    }
#pragma unroll
    for (int j = 0; j < 4; ++j) {
      const int n = wave_n * 64 + j * 16 + nb;  // B: lane = N col, K 0-7/8-15 + 16-23/24-31 by hl
      v8bf lo = *(const v8bf*)&lds_b[cur][n * 40 + hl * 8];
      v8bf hi = *(const v8bf*)&lds_b[cur][n * 40 + 16 + hl * 8];
      bfrag[j] = pack16(lo, hi);
    }
#pragma unroll
    for (int i = 0; i < 2; ++i)
#pragma unroll
      for (int j = 0; j < 4; ++j)
        acc[i][j] = wmma_bf16(afrag[i], bfrag[j], acc[i][j]);

    wait_async0();   // next tile landed in LDS
    __syncthreads(); // all waves done reading `cur` before it is refilled
  }

#pragma unroll
  for (int i = 0; i < 2; ++i)
#pragma unroll
    for (int j = 0; j < 4; ++j) {
      const int col = n_block * 128 + wave_n * 64 + j * 16 + nb;
      const float bv = bias[col];
#pragma unroll
      for (int r = 0; r < 8; ++r) {
        const int row = m_block * 128 + wave_m * 32 + i * 16 + hl * 8 + r;
        const float v = acc[i][j][r] + bv;
        if (OUT_BF16)
          ((bf16*)outp)[(size_t)row * N + col] = (bf16)v;
        else
          ((float*)outp)[(size_t)row * N + col] = v;
      }
    }
}

// ------------------------------- per-head RMSNorm over D=128 + split to (B,h,S,D)
__global__ __launch_bounds__(256) void rmsnorm_split(const bf16* __restrict__ in,
                                                     const float* __restrict__ gamma,
                                                     bf16* __restrict__ out,
                                                     int nheads) {
  const int lane = threadIdx.x & 31, w = threadIdx.x >> 5;
  const size_t vec = (size_t)blockIdx.x * 8 + w;  // over B*S*nheads
  const int h = (int)(vec % nheads);
  const size_t bs = vec / nheads;
  const int s = (int)(bs % CS);
  const int b = (int)(bs / CS);

  const bf16* src = in + bs * (size_t)(nheads * CD) + (size_t)h * CD;
  float v[4];
  float ss = 0.f;
#pragma unroll
  for (int e = 0; e < 4; ++e) {
    v[e] = (float)src[lane * 4 + e];
    ss += v[e] * v[e];
  }
  ss = full_sum32(ss);
  const float inv = rsqrtf(ss * (1.0f / CD) + CEPS);
  bf16* dst = out + (((size_t)b * nheads + h) * CS + s) * CD;
#pragma unroll
  for (int e = 0; e < 4; ++e)
    dst[lane * 4 + e] = (bf16)(v[e] * inv * gamma[lane * 4 + e]);
}

// ------------------------------------------- V relayout (B,S,KV,D) -> (B,KV,S,D)
__global__ void v_split(const bf16* __restrict__ in, bf16* __restrict__ out) {
  const size_t total = (size_t)CB * CS * CKV * (CD / 8);
  size_t idx = (size_t)blockIdx.x * blockDim.x + threadIdx.x;
  if (idx >= total) return;
  const int dc = (int)(idx % (CD / 8));
  size_t r = idx / (CD / 8);
  const int kvh = (int)(r % CKV); r /= CKV;
  const int s = (int)(r % CS);
  const int b = (int)(r / CS);
  v8bf val = *(const v8bf*)(in + (((size_t)b * CS + s) * CKV + kvh) * CD + dc * 8);
  *(v8bf*)(out + (((size_t)b * CKV + kvh) * CS + s) * CD + dc * 8) = val;
}

// --------------------------------------------------- causal flash attention (GQA)
// grid (S/128, H, B); 8 waves; wave owns 16 query rows; K/V streamed 64 at a time.
// K tile arrives via async-to-LDS; V is register-transposed into LDS.
__global__ __launch_bounds__(256) void flash_attn(const bf16* __restrict__ Qh,
                                                  const bf16* __restrict__ Kh,
                                                  const bf16* __restrict__ Vh,
                                                  bf16* __restrict__ Ob) {
  __shared__ bf16 lds_k[64 * 136];      // [kv][d]
  __shared__ bf16 lds_vt[128 * 72];     // [d][kv] (transposed for B operand)
  __shared__ bf16 lds_p[8 * 16 * 72];   // per-wave P staging [16][64+pad]
  const int qb = blockIdx.x, h = blockIdx.y, b = blockIdx.z;
  const int kvh = h / CG;
  const int t = threadIdx.x, lane = t & 31, w = t >> 5;
  const int nb = lane & 15, hl = lane >> 4;

  // Preload this wave's Q rows as A fragments (4 k-steps of 32 over D=128).
  v16bf qa[4];
  {
    const int m = qb * 128 + w * 16 + nb;
    const bf16* qsrc = Qh + (((size_t)b * CH + h) * CS + m) * CD;
#pragma unroll
    for (int kk = 0; kk < 4; ++kk) {
      v8bf lo = *(const v8bf*)(qsrc + kk * 32 + hl * 8);
      v8bf hi = *(const v8bf*)(qsrc + kk * 32 + 16 + hl * 8);
      qa[kk] = pack16(lo, hi);
    }
  }

  v8f o[8] = {};
  float mrow[8], lrow[8];
#pragma unroll
  for (int r = 0; r < 8; ++r) { mrow[r] = -__builtin_inff(); lrow[r] = 0.f; }

  const int rowg0 = qb * 128 + w * 16 + hl * 8;
  const int kv_end = qb * 128 + 128;

  for (int kv0 = 0; kv0 < kv_end; kv0 += 64) {
    {  // cooperative K/V tile load
      const int row = t >> 2, c = (t & 3) * 32;
      const bf16* ks = Kh + (((size_t)b * CKV + kvh) * CS + kv0 + row) * CD + c;
      const bf16* vs = Vh + (((size_t)b * CKV + kvh) * CS + kv0 + row) * CD + c;
      bf16* lk = &lds_k[row * 136 + c];
#pragma unroll
      for (int q4 = 0; q4 < 4; ++q4)          // K: async global -> LDS, no VGPRs
        async_ld_b128(lk + q4 * 8, ks + q4 * 8);
#pragma unroll
      for (int q4 = 0; q4 < 4; ++q4) {        // V: transpose through registers
        v8bf vv = *(const v8bf*)(vs + q4 * 8);
#pragma unroll
        for (int e = 0; e < 8; ++e) lds_vt[(c + q4 * 8 + e) * 72 + row] = vv[e];
      }
    }
    wait_async0();
    __syncthreads();

    // S = Q @ K^T  (4 n-tiles of 16 kv cols, 4 k-steps over D)
    v8f sfr[4] = {};
#pragma unroll
    for (int nt = 0; nt < 4; ++nt) {
      const int col = nt * 16 + nb;
#pragma unroll
      for (int kk = 0; kk < 4; ++kk) {
        v8bf lo = *(const v8bf*)&lds_k[col * 136 + kk * 32 + hl * 8];
        v8bf hi = *(const v8bf*)&lds_k[col * 136 + kk * 32 + 16 + hl * 8];
        sfr[nt] = wmma_bf16(qa[kk], pack16(lo, hi), sfr[nt]);
      }
    }

    // scale + causal mask
#pragma unroll
    for (int nt = 0; nt < 4; ++nt)
#pragma unroll
      for (int r = 0; r < 8; ++r) {
        float sv = sfr[nt][r] * CSCALE;
        if (kv0 + nt * 16 + nb > rowg0 + r) sv = -__builtin_inff();
        sfr[nt][r] = sv;
      }

    // online softmax
    float nmax[8], alpha[8], rsum[8];
#pragma unroll
    for (int r = 0; r < 8; ++r) {
      float tm = fmaxf(fmaxf(sfr[0][r], sfr[1][r]), fmaxf(sfr[2][r], sfr[3][r]));
      tm = half_max16(tm);
      nmax[r] = fmaxf(mrow[r], tm);
      alpha[r] = __expf(mrow[r] - nmax[r]);
      mrow[r] = nmax[r];
      rsum[r] = 0.f;
    }
#pragma unroll
    for (int nt = 0; nt < 4; ++nt)
#pragma unroll
      for (int r = 0; r < 8; ++r) {
        const float p = __expf(sfr[nt][r] - nmax[r]);
        rsum[r] += p;
        lds_p[(w * 16 + hl * 8 + r) * 72 + nt * 16 + nb] = (bf16)p;
      }
#pragma unroll
    for (int r = 0; r < 8; ++r) {
      lrow[r] = lrow[r] * alpha[r] + half_sum16(rsum[r]);
#pragma unroll
      for (int dt = 0; dt < 8; ++dt) o[dt][r] *= alpha[r];
    }

    asm volatile("s_wait_dscnt 0" ::: "memory");  // P stores -> A-frag reloads

    // O += P @ V  (8 d-tiles, 2 k-steps of 32 kv)
#pragma unroll
    for (int k2 = 0; k2 < 2; ++k2) {
      v8bf plo = *(const v8bf*)&lds_p[(w * 16 + nb) * 72 + k2 * 32 + hl * 8];
      v8bf phi = *(const v8bf*)&lds_p[(w * 16 + nb) * 72 + k2 * 32 + 16 + hl * 8];
      v16bf ap = pack16(plo, phi);
#pragma unroll
      for (int dt = 0; dt < 8; ++dt) {
        const int dcol = dt * 16 + nb;
        v8bf lo = *(const v8bf*)&lds_vt[dcol * 72 + k2 * 32 + hl * 8];
        v8bf hi = *(const v8bf*)&lds_vt[dcol * 72 + k2 * 32 + 16 + hl * 8];
        o[dt] = wmma_bf16(ap, pack16(lo, hi), o[dt]);
      }
    }
    __syncthreads();
  }

  // normalize + write merged heads (B,S,E) bf16
#pragma unroll
  for (int dt = 0; dt < 8; ++dt)
#pragma unroll
    for (int r = 0; r < 8; ++r) {
      const int row = qb * 128 + w * 16 + hl * 8 + r;
      const int col = h * CD + dt * 16 + nb;
      Ob[((size_t)b * CS + row) * CE + col] = (bf16)(o[dt][r] / lrow[r]);
    }
}

// --------------------------------------------------------------------------------
extern "C" void kernel_launch(void* const* d_in, const int* in_sizes, int n_in,
                              void* d_out, int out_size, void* d_ws, size_t ws_size,
                              hipStream_t stream) {
  const float* x  = (const float*)d_in[0];
  const float* Wq = (const float*)d_in[1];
  const float* bq = (const float*)d_in[2];
  const float* Wk = (const float*)d_in[3];
  const float* bk = (const float*)d_in[4];
  const float* Wv = (const float*)d_in[5];
  const float* bv = (const float*)d_in[6];
  const float* Wo = (const float*)d_in[7];
  const float* bo = (const float*)d_in[8];
  const float* qn = (const float*)d_in[9];
  const float* kn = (const float*)d_in[10];
  float* out = (float*)d_out;

  char* ws = (char*)d_ws;
  auto carve = [&](size_t elems) {
    bf16* p = (bf16*)ws;
    ws += (elems * sizeof(bf16) + 255) & ~(size_t)255;
    return p;
  };
  const size_t nX  = (size_t)CB * CS * CE;       // 8M
  const size_t nWq = (size_t)CE * CE;            // 4M
  const size_t nWk = (size_t)CKV * CD * CE;      // 2M
  bf16* x_bf   = carve(nX);
  bf16* wq_bf  = carve(nWq);
  bf16* wk_bf  = carve(nWk);
  bf16* wv_bf  = carve(nWk);
  bf16* wo_bf  = carve(nWq);
  bf16* proj_q = carve(nX);
  bf16* proj_k = carve((size_t)CB * CS * CKV * CD);
  bf16* proj_v = carve((size_t)CB * CS * CKV * CD);
  bf16* q_hd   = carve((size_t)CB * CH * CS * CD);
  bf16* k_hd   = carve((size_t)CB * CKV * CS * CD);
  bf16* v_hd   = carve((size_t)CB * CKV * CS * CD);
  bf16* attn_bf = carve(nX);

  cast_f32_bf16<<<2048, 256, 0, stream>>>(x,  x_bf,  (int)nX);
  cast_f32_bf16<<<2048, 256, 0, stream>>>(Wq, wq_bf, (int)nWq);
  cast_f32_bf16<<<1024, 256, 0, stream>>>(Wk, wk_bf, (int)nWk);
  cast_f32_bf16<<<1024, 256, 0, stream>>>(Wv, wv_bf, (int)nWk);
  cast_f32_bf16<<<2048, 256, 0, stream>>>(Wo, wo_bf, (int)nWq);

  const int M = CB * CS;  // 4096
  gemm_bias<true><<<dim3(CE / 128, M / 128), 256, 0, stream>>>(
      x_bf, wq_bf, bq, proj_q, M, CE, CE);
  gemm_bias<true><<<dim3(CKV * CD / 128, M / 128), 256, 0, stream>>>(
      x_bf, wk_bf, bk, proj_k, M, CKV * CD, CE);
  gemm_bias<true><<<dim3(CKV * CD / 128, M / 128), 256, 0, stream>>>(
      x_bf, wv_bf, bv, proj_v, M, CKV * CD, CE);

  rmsnorm_split<<<CB * CS * CH / 8, 256, 0, stream>>>(proj_q, qn, q_hd, CH);
  rmsnorm_split<<<CB * CS * CKV / 8, 256, 0, stream>>>(proj_k, kn, k_hd, CKV);
  v_split<<<(CB * CS * CKV * CD / 8 + 255) / 256, 256, 0, stream>>>(proj_v, v_hd);

  flash_attn<<<dim3(CS / 128, CH, CB), 256, 0, stream>>>(q_hd, k_hd, v_hd, attn_bf);

  gemm_bias<false><<<dim3(CE / 128, M / 128), 256, 0, stream>>>(
      attn_bf, wo_bf, bo, out, M, CE, CE);
}
